// FluxJointAttention_16716012716262
// MI455X (gfx1250) — matrix operand order repaired and checked by reading, not compiled, not run on hardware
//
#include <hip/hip_runtime.h>

#define DIMN   3072
#define HEADS  24
#define HD     128
#define BATCH  2
#define S_IMG  1024
#define S_TXT  512
#define SEQ    1536
#define EPSF   1e-6f
#define SCALE  0.08838834764831845f  /* 1/sqrt(128) */

typedef _Float16 h16;
typedef __attribute__((ext_vector_type(16))) _Float16 v16h;
typedef __attribute__((ext_vector_type(8)))  float    v8f;

union H16 { v16h v; uint4 u[2]; };
union H8  { uint4 u; h16 h[8]; };
union H2  { unsigned int u; h16 h[2]; };

static __device__ __forceinline__ v8f wmma_f16(v16h a, v16h b, v8f c) {
  // D = A(16x32 f16) * B(32x16 f16) + C(16x16 f32)
  return __builtin_amdgcn_wmma_f32_16x16x32_f16(false, a, false, b, (short)0, c, false, false);
}

// Async DMA: global -> LDS, 16B per lane, tracked by ASYNCcnt (CDNA5 TDM-lite path).
// VDST operand = LDS byte address (low 32 bits of generic pointer to __shared__).
static __device__ __forceinline__ void async_ld_b128(const void* g, void* l) {
  asm volatile("global_load_async_to_lds_b128 %0, %1, off"
               :: "v"((unsigned)(unsigned long long)l), "v"(g)
               : "memory");
}
static __device__ __forceinline__ void wait_async0() {
  asm volatile("s_wait_asynccnt 0" ::: "memory");
}

// ---------------------------------------------------------------- f32 -> f16
__global__ __launch_bounds__(256) void cvt_kernel(const float* __restrict__ s,
                                                  h16* __restrict__ d, int n) {
  int i = (blockIdx.x * 256 + threadIdx.x) * 4;
  if (i < n) {
    float4 f = *(const float4*)(s + i);
    union { uint2 u; h16 h[4]; } o;
    o.h[0] = (h16)f.x; o.h[1] = (h16)f.y; o.h[2] = (h16)f.z; o.h[3] = (h16)f.w;
    *(uint2*)(d + i) = o.u;
  }
}

// ------------------------------------------------- fused QKV GEMM + RMS+RoPE
struct GemmBufs { h16 A[64 * 32]; h16 Bt[128 * 40]; };
union SmemQKV { GemmBufs g; float C[64 * 128]; };

__global__ __launch_bounds__(256) void qkv_kernel(
    const h16* __restrict__ x16, const h16* __restrict__ ctx16,
    const h16* __restrict__ w16,
    const float* __restrict__ bq,  const float* __restrict__ bk,  const float* __restrict__ bv,
    const float* __restrict__ bqc, const float* __restrict__ bkc, const float* __restrict__ bvc,
    const float* __restrict__ gq,  const float* __restrict__ gk,
    const float* __restrict__ gqc, const float* __restrict__ gkc,
    const float* __restrict__ rope_cos, const float* __restrict__ rope_sin,
    h16* __restrict__ qj, h16* __restrict__ kj, h16* __restrict__ vj) {
  __shared__ SmemQKV sm;

  const int tid  = threadIdx.x;
  const int lane = tid & 31, wave = tid >> 5;
  const int wm = wave & 1, wn = wave >> 1;
  const int hh = lane >> 4, ln = lane & 15;

  const int rt = blockIdx.x, head = blockIdx.y, which = blockIdx.z;
  const int r0 = rt * 64;
  const int b  = r0 / SEQ;
  const int s0 = r0 - b * SEQ;               // joint seq offset of tile
  const bool is_ctx = (s0 < S_TXT);
  const int n0 = head * HD;

  const h16* in_base = is_ctx ? (ctx16 + (size_t)(b * S_TXT + s0) * DIMN)
                              : (x16  + (size_t)(b * S_IMG + (s0 - S_TXT)) * DIMN);
  const h16* W = w16 + (size_t)(which + (is_ctx ? 3 : 0)) * (size_t)DIMN * DIMN;

  v8f acc[2][2] = {};
  const int arow = tid >> 2, ach = tid & 3;

  for (int kt = 0; kt < DIMN / 32; ++kt) {
    const int k0 = kt * 32;
    // B tile: global -> regs (reads can overlap the barrier)
    uint4 bR[2];
#pragma unroll
    for (int i = 0; i < 2; ++i) {
      int idx = tid + i * 256, kk = idx >> 4, nc = idx & 15;
      bR[i] = *(const uint4*)(W + (size_t)(k0 + kk) * DIMN + n0 + nc * 8);
    }
    __builtin_prefetch(W + (size_t)(k0 + 32 + (tid >> 4)) * DIMN + n0, 0, 1);

    __syncthreads();  // prior iteration's fragment reads retired
    // A tile: async DMA global -> LDS (bypasses VGPRs, ASYNCcnt-tracked)
    async_ld_b128(in_base + (size_t)arow * DIMN + k0 + ach * 8,
                  &sm.g.A[arow * 32 + ach * 8]);
    // B tile: transpose-scatter into LDS
#pragma unroll
    for (int i = 0; i < 2; ++i) {
      int idx = tid + i * 256, kk = idx >> 4, nc = idx & 15;
      H8 t; t.u = bR[i];
#pragma unroll
      for (int c = 0; c < 8; ++c) sm.g.Bt[(nc * 8 + c) * 40 + kk] = t.h[c];
    }
    wait_async0();
    __syncthreads();

    H16 af[2], bf[2];
#pragma unroll
    for (int mt = 0; mt < 2; ++mt) {
      int m = wm * 32 + mt * 16 + ln;
      af[mt].u[0] = *(const uint4*)(&sm.g.A[m * 32 + 8 * hh]);
      af[mt].u[1] = *(const uint4*)(&sm.g.A[m * 32 + 16 + 8 * hh]);
    }
#pragma unroll
    for (int nt = 0; nt < 2; ++nt) {
      int col = wn * 32 + nt * 16 + ln;
      bf[nt].u[0] = *(const uint4*)(&sm.g.Bt[col * 40 + 8 * hh]);
      bf[nt].u[1] = *(const uint4*)(&sm.g.Bt[col * 40 + 16 + 8 * hh]);
    }
#pragma unroll
    for (int mt = 0; mt < 2; ++mt)
#pragma unroll
      for (int nt = 0; nt < 2; ++nt)
        acc[mt][nt] = wmma_f16(af[mt].v, bf[nt].v, acc[mt][nt]);
  }

  __syncthreads();
#pragma unroll
  for (int mt = 0; mt < 2; ++mt)
#pragma unroll
    for (int nt = 0; nt < 2; ++nt) {
      int col = wn * 32 + nt * 16 + ln;
#pragma unroll
      for (int r = 0; r < 8; ++r) {
        int row = wm * 32 + mt * 16 + r + 8 * hh;
        sm.C[row * 128 + col] = acc[mt][nt][r];
      }
    }
  __syncthreads();

  if (tid < 64) {
    const int row = tid;
    const int sj  = s0 + row;   // joint position (ctx first -> matches rope pos)
    float* rowp = &sm.C[row * 128];
    const float* bias = is_ctx ? (which == 0 ? bqc : (which == 1 ? bkc : bvc))
                               : (which == 0 ? bq  : (which == 1 ? bk  : bv));
    if (which == 2) {
      h16* dst = vj + ((size_t)(b * HEADS + head) * SEQ + sj) * HD;
#pragma unroll
      for (int d0 = 0; d0 < HD; d0 += 8) {
        H8 o;
#pragma unroll
        for (int c = 0; c < 8; ++c) o.h[c] = (h16)(rowp[d0 + c] + bias[n0 + d0 + c]);
        *(uint4*)(dst + d0) = o.u;
      }
    } else {
      const float* gain = (which == 0) ? (is_ctx ? gqc : gq) : (is_ctx ? gkc : gk);
      float ss = 0.f;
      for (int d = 0; d < HD; ++d) {
        float v = rowp[d] + bias[n0 + d];
        ss += v * v;
        rowp[d] = v;
      }
      const float rms = rsqrtf(ss * (1.0f / HD) + EPSF);
      h16* dst = (which == 0 ? qj : kj) + ((size_t)(b * HEADS + head) * SEQ + sj) * HD;
      for (int i = 0; i < HD / 2; ++i) {
        float x1 = rowp[2 * i]     * rms * gain[2 * i];
        float x2 = rowp[2 * i + 1] * rms * gain[2 * i + 1];
        float c = rope_cos[sj * (HD / 2) + i];
        float s = rope_sin[sj * (HD / 2) + i];
        H2 o;
        o.h[0] = (h16)(x1 * c - x2 * s);
        o.h[1] = (h16)(x1 * s + x2 * c);
        *(unsigned int*)(dst + 2 * i) = o.u;
      }
    }
  }
}

// -------------------------------------------------------- flash attention
__global__ __launch_bounds__(128) void attn_kernel(
    const h16* __restrict__ qj, const h16* __restrict__ kj,
    const h16* __restrict__ vj, h16* __restrict__ ao) {
  __shared__ h16 Ksh[32 * 128];   // K chunk, [kv][d]
  __shared__ h16 Vsh[128 * 40];   // V chunk transposed, [d][kv] (pad 40)
  __shared__ h16 Psh[4 * 16 * 32];// per-wave P re-layout patch

  const int tid = threadIdx.x, lane = tid & 31, wave = tid >> 5;
  const int hh = lane >> 4, ln = lane & 15;
  const int qt = blockIdx.x, head = blockIdx.y, b = blockIdx.z;
  const size_t bh = (size_t)(b * HEADS + head);
  const h16* qb = qj + bh * SEQ * HD;
  const h16* kb = kj + bh * SEQ * HD;
  const h16* vb = vj + bh * SEQ * HD;
  const int sq0 = qt * 64 + wave * 16;

  // Q 16x128 held as four A fragments (k-chunks of 32)
  H16 qf[4];
#pragma unroll
  for (int c = 0; c < 4; ++c) {
    const h16* p = qb + (size_t)(sq0 + ln) * HD + c * 32 + 8 * hh;
    qf[c].u[0] = *(const uint4*)(p);
    qf[c].u[1] = *(const uint4*)(p + 16);
  }

  v8f of[8] = {};
  float mi[8], li[8];
#pragma unroll
  for (int r = 0; r < 8; ++r) { mi[r] = -3.0e38f; li[r] = 0.f; }

  for (int kt = 0; kt < SEQ / 32; ++kt) {
    const int kv0 = kt * 32;
    __syncthreads();  // prior iteration's K/V fragment reads retired
#pragma unroll
    for (int i = 0; i < 4; ++i) {
      int idx = tid + i * 128;
      int row = idx >> 4, ch = idx & 15;
      // K chunk: async DMA straight into LDS
      async_ld_b128(kb + (size_t)(kv0 + row) * HD + ch * 8,
                    &Ksh[row * 128 + ch * 8]);
      // V chunk: needs transpose -> manual scatter
      H8 t; t.u = *(const uint4*)(vb + (size_t)(kv0 + row) * HD + ch * 8);
#pragma unroll
      for (int c = 0; c < 8; ++c) Vsh[(ch * 8 + c) * 40 + row] = t.h[c];
    }
    wait_async0();
    __syncthreads();

    // scores: Q(16x128) . K_chunk^T(128x32) -> two 16x16 C fragments
    v8f sc[2] = {};
#pragma unroll
    for (int nt = 0; nt < 2; ++nt) {
      int n = nt * 16 + ln;
#pragma unroll
      for (int c = 0; c < 4; ++c) {
        H16 bfr;
        bfr.u[0] = *(const uint4*)(&Ksh[n * 128 + c * 32 + 8 * hh]);
        bfr.u[1] = *(const uint4*)(&Ksh[n * 128 + c * 32 + 16 + 8 * hh]);
        sc[nt] = wmma_f16(qf[c].v, bfr.v, sc[nt]);
      }
    }

    // online softmax (row = r + 8*hh, cols across lanes 0..15 of each half)
#pragma unroll
    for (int r = 0; r < 8; ++r) {
      float s0v = sc[0][r] * SCALE, s1v = sc[1][r] * SCALE;
      float mx = fmaxf(s0v, s1v);
#pragma unroll
      for (int off = 1; off < 16; off <<= 1)
        mx = fmaxf(mx, __shfl_xor(mx, off, 32));
      float mnew  = fmaxf(mi[r], mx);
      float alpha = __expf(mi[r] - mnew);
      float p0 = __expf(s0v - mnew), p1 = __expf(s1v - mnew);
      float rs = p0 + p1;
#pragma unroll
      for (int off = 1; off < 16; off <<= 1)
        rs += __shfl_xor(rs, off, 32);
      li[r] = li[r] * alpha + rs;
      mi[r] = mnew;
#pragma unroll
      for (int t = 0; t < 8; ++t) of[t][r] = of[t][r] * alpha;
      int m = r + 8 * hh;
      Psh[wave * 512 + m * 32 + ln]      = (h16)p0;
      Psh[wave * 512 + m * 32 + 16 + ln] = (h16)p1;
    }

    // P(16x32) as A fragment (per-wave LDS patch, wave-local in-order DS)
    H16 pf;
    {
      const h16* pp = &Psh[wave * 512 + ln * 32 + 8 * hh];
      pf.u[0] = *(const uint4*)(pp);
      pf.u[1] = *(const uint4*)(pp + 16);
    }
    // O += P . V_chunk  (V transposed in LDS -> contiguous B fragments)
#pragma unroll
    for (int nt2 = 0; nt2 < 8; ++nt2) {
      int d = nt2 * 16 + ln;
      H16 vf;
      vf.u[0] = *(const uint4*)(&Vsh[d * 40 + 8 * hh]);
      vf.u[1] = *(const uint4*)(&Vsh[d * 40 + 16 + 8 * hh]);
      of[nt2] = wmma_f16(pf.v, vf.v, of[nt2]);
    }
  }

#pragma unroll
  for (int r = 0; r < 8; ++r) {
    float inv = 1.0f / li[r];
    int srow = sq0 + r + 8 * hh;
    h16* dst = ao + ((size_t)(b * SEQ + srow)) * DIMN + head * HD;
#pragma unroll
    for (int t = 0; t < 8; ++t)
      dst[t * 16 + ln] = (h16)(of[t][r] * inv);
  }
}

// ------------------------------------------------------ output projection
__global__ __launch_bounds__(256) void out_kernel(
    const h16* __restrict__ ao, const h16* __restrict__ w16,
    const float* __restrict__ b_out, const float* __restrict__ b_add_out,
    float* __restrict__ out) {
  __shared__ h16 A[64 * 32];
  __shared__ h16 Bt[128 * 40];

  const int tid  = threadIdx.x;
  const int lane = tid & 31, wave = tid >> 5;
  const int wm = wave & 1, wn = wave >> 1;
  const int hh = lane >> 4, ln = lane & 15;
  const int rt = blockIdx.x, nblk = blockIdx.y;
  const int n0 = nblk * 128;
  const int r0 = rt * 64;
  const int b  = r0 / SEQ, s0 = r0 - b * SEQ;
  const bool is_txt = (s0 < S_TXT);

  const h16* in_base = ao + (size_t)(b * SEQ + s0) * DIMN;
  const h16* W = w16 + (size_t)(is_txt ? 7 : 6) * (size_t)DIMN * DIMN;
  const float* bias = is_txt ? b_add_out : b_out;
  float* dst_base = is_txt
      ? (out + (size_t)(b * S_TXT + s0) * DIMN)
      : (out + (size_t)BATCH * S_TXT * DIMN + (size_t)(b * S_IMG + (s0 - S_TXT)) * DIMN);

  v8f acc[2][2] = {};
  const int arow = tid >> 2, ach = tid & 3;

  for (int kt = 0; kt < DIMN / 32; ++kt) {
    const int k0 = kt * 32;
    uint4 bR[2];
#pragma unroll
    for (int i = 0; i < 2; ++i) {
      int idx = tid + i * 256, kk = idx >> 4, nc = idx & 15;
      bR[i] = *(const uint4*)(W + (size_t)(k0 + kk) * DIMN + n0 + nc * 8);
    }
    __builtin_prefetch(W + (size_t)(k0 + 32 + (tid >> 4)) * DIMN + n0, 0, 1);

    __syncthreads();
    async_ld_b128(in_base + (size_t)arow * DIMN + k0 + ach * 8,
                  &A[arow * 32 + ach * 8]);
#pragma unroll
    for (int i = 0; i < 2; ++i) {
      int idx = tid + i * 256, kk = idx >> 4, nc = idx & 15;
      H8 t; t.u = bR[i];
#pragma unroll
      for (int c = 0; c < 8; ++c) Bt[(nc * 8 + c) * 40 + kk] = t.h[c];
    }
    wait_async0();
    __syncthreads();

    H16 af[2], bf[2];
#pragma unroll
    for (int mt = 0; mt < 2; ++mt) {
      int m = wm * 32 + mt * 16 + ln;
      af[mt].u[0] = *(const uint4*)(&A[m * 32 + 8 * hh]);
      af[mt].u[1] = *(const uint4*)(&A[m * 32 + 16 + 8 * hh]);
    }
#pragma unroll
    for (int nt = 0; nt < 2; ++nt) {
      int col = wn * 32 + nt * 16 + ln;
      bf[nt].u[0] = *(const uint4*)(&Bt[col * 40 + 8 * hh]);
      bf[nt].u[1] = *(const uint4*)(&Bt[col * 40 + 16 + 8 * hh]);
    }
#pragma unroll
    for (int mt = 0; mt < 2; ++mt)
#pragma unroll
      for (int nt = 0; nt < 2; ++nt)
        acc[mt][nt] = wmma_f16(af[mt].v, bf[nt].v, acc[mt][nt]);
  }

#pragma unroll
  for (int mt = 0; mt < 2; ++mt)
#pragma unroll
    for (int nt = 0; nt < 2; ++nt) {
      int col = wn * 32 + nt * 16 + ln;
      float bv = bias[n0 + col];
#pragma unroll
      for (int r = 0; r < 8; ++r) {
        int row = wm * 32 + mt * 16 + r + 8 * hh;
        dst_base[(size_t)row * DIMN + n0 + col] = acc[mt][nt][r] + bv;
      }
    }
}

// ---------------------------------------------------------------- launcher
extern "C" void kernel_launch(void* const* d_in, const int* in_sizes, int n_in,
                              void* d_out, int out_size, void* d_ws, size_t ws_size,
                              hipStream_t stream) {
  (void)in_sizes; (void)n_in; (void)out_size; (void)ws_size;

  const float* x        = (const float*)d_in[0];
  const float* ctx      = (const float*)d_in[1];
  const float* rope_cos = (const float*)d_in[2];
  const float* rope_sin = (const float*)d_in[3];
  const float* wq  = (const float*)d_in[4];
  const float* bq  = (const float*)d_in[5];
  const float* wk  = (const float*)d_in[6];
  const float* bk  = (const float*)d_in[7];
  const float* wv  = (const float*)d_in[8];
  const float* bv  = (const float*)d_in[9];
  const float* wqc = (const float*)d_in[10];
  const float* bqc = (const float*)d_in[11];
  const float* wkc = (const float*)d_in[12];
  const float* bkc = (const float*)d_in[13];
  const float* wvc = (const float*)d_in[14];
  const float* bvc = (const float*)d_in[15];
  const float* w_out     = (const float*)d_in[16];
  const float* b_out     = (const float*)d_in[17];
  const float* w_add_out = (const float*)d_in[18];
  const float* b_add_out = (const float*)d_in[19];
  const float* g_q  = (const float*)d_in[20];
  const float* g_k  = (const float*)d_in[21];
  const float* g_qc = (const float*)d_in[22];
  const float* g_kc = (const float*)d_in[23];

  const size_t DD = (size_t)DIMN * DIMN;
  h16* w16   = (h16*)d_ws;                                  // 8 * DD halves
  h16* x16   = w16 + 8 * DD;
  h16* ctx16 = x16 + (size_t)BATCH * S_IMG * DIMN;
  h16* qjp   = ctx16 + (size_t)BATCH * S_TXT * DIMN;
  h16* kjp   = qjp + (size_t)BATCH * SEQ * DIMN;
  h16* vjp   = kjp + (size_t)BATCH * SEQ * DIMN;
  h16* aop   = vjp + (size_t)BATCH * SEQ * DIMN;

  auto cvt = [&](const float* s, h16* d, size_t n) {
    int blocks = (int)((n / 4 + 255) / 256);
    cvt_kernel<<<blocks, 256, 0, stream>>>(s, d, (int)n);
  };
  cvt(wq,        w16 + 0 * DD, DD);
  cvt(wk,        w16 + 1 * DD, DD);
  cvt(wv,        w16 + 2 * DD, DD);
  cvt(wqc,       w16 + 3 * DD, DD);
  cvt(wkc,       w16 + 4 * DD, DD);
  cvt(wvc,       w16 + 5 * DD, DD);
  cvt(w_out,     w16 + 6 * DD, DD);
  cvt(w_add_out, w16 + 7 * DD, DD);
  cvt(x,   x16,   (size_t)BATCH * S_IMG * DIMN);
  cvt(ctx, ctx16, (size_t)BATCH * S_TXT * DIMN);

  qkv_kernel<<<dim3((BATCH * SEQ) / 64, HEADS, 3), 256, 0, stream>>>(
      x16, ctx16, w16, bq, bk, bv, bqc, bkc, bvc,
      g_q, g_k, g_qc, g_kc, rope_cos, rope_sin, qjp, kjp, vjp);

  attn_kernel<<<dim3(SEQ / 64, HEADS, BATCH), 128, 0, stream>>>(qjp, kjp, vjp, aop);

  out_kernel<<<dim3((BATCH * SEQ) / 64, HEADS, 1), 256, 0, stream>>>(
      aop, w16, b_out, b_add_out, (float*)d_out);
}
